// EncoderBlock_49924699849257
// MI455X (gfx1250) — matrix operand order repaired and checked by reading, not compiled, not run on hardware
//
#include <hip/hip_runtime.h>

#define DM   768
#define NH   12
#define DKH  64
#define DFF  3072
#define BBAT 8
#define SSEQ 1024
#define MROWS (BBAT * SSEQ)

typedef __attribute__((ext_vector_type(8)))  _Float16 v8h;
typedef __attribute__((ext_vector_type(16))) _Float16 v16h;
typedef __attribute__((ext_vector_type(8)))  float    v8f;

#define WMMA_F16(a, b, c) \
  __builtin_amdgcn_wmma_f32_16x16x32_f16(false, (a), false, (b), (short)0, (c), false, false)

// A-matrix fragment (16x32 f16): lane holds row L%16; halfs 0..7 = K h*8..h*8+7,
// halfs 8..15 = K 16+h*8..16+h*8+7  (ISA 7.12.2, 16-bit A 16x32 table)
static __device__ __forceinline__ v16h frag_ld_a(const _Float16* p, int hf) {
  v8h lo = *reinterpret_cast<const v8h*>(p + hf * 8);
  v8h hi = *reinterpret_cast<const v8h*>(p + 16 + hf * 8);
  v16h r;
#pragma unroll
  for (int i = 0; i < 8; ++i) { r[i] = lo[i]; r[i + 8] = hi[i]; }
  return r;
}

// B-matrix fragment (32x16 f16): lane holds col L%16; halfs 0..7 = K h*8..h*8+7,
// halfs 8..15 = K 16+h*8..16+h*8+7 (same k order as frag_ld_a)
static __device__ __forceinline__ v16h frag_ld_b(const _Float16* p, int hf) {
  v8h lo = *reinterpret_cast<const v8h*>(p + hf * 8);
  v8h hi = *reinterpret_cast<const v8h*>(p + 16 + hf * 8);
  v16h r;
#pragma unroll
  for (int i = 0; i < 8; ++i) { r[i] = lo[i]; r[i + 8] = hi[i]; }
  return r;
}

// Async DMA of one 16-byte chunk of the shared A tile (CDNA5 ASYNCcnt path).
// LDS aperture occupies addr[63:32] only, so the low 32 bits of a generic
// pointer into __shared__ are the DS byte address (ISA 10.2).
static __device__ __forceinline__ void async_cp_b128(const _Float16* gp,
                                                     _Float16* lp) {
  const unsigned lds = (unsigned)(unsigned long long)(uintptr_t)lp;
  asm volatile("global_load_async_to_lds_b128 %0, %1, off"
               :: "v"(lds), "v"((unsigned long long)(uintptr_t)gp)
               : "memory");
}

// ---------------- fp32 -> f16 weight conversion ----------------
__global__ void cvt_f32_f16_kernel(const float* __restrict__ in,
                                   _Float16* __restrict__ out, int n) {
  int i = blockIdx.x * blockDim.x + threadIdx.x;
  if (i < n) out[i] = (_Float16)in[i];
}

// ---------------- LayerNorm (Bessel-corrected std, std+eps denom) ----------
__global__ __launch_bounds__(256) void ln_kernel(const float* __restrict__ x,
                                                 const float* __restrict__ alpha,
                                                 const float* __restrict__ beta,
                                                 _Float16* __restrict__ out,
                                                 int rows) {
  const int row  = blockIdx.x * (blockDim.x >> 5) + (threadIdx.x >> 5);
  const int lane = threadIdx.x & 31;
  if (row >= rows) return;
  const float* xr = x + (size_t)row * DM;
  float s = 0.f;
  for (int i = lane; i < DM; i += 32) s += xr[i];
#pragma unroll
  for (int off = 16; off; off >>= 1) s += __shfl_xor(s, off, 32);
  const float mean = s * (1.0f / DM);
  float v = 0.f;
  for (int i = lane; i < DM; i += 32) { float d = xr[i] - mean; v += d * d; }
#pragma unroll
  for (int off = 16; off; off >>= 1) v += __shfl_xor(v, off, 32);
  const float stdv = sqrtf(v / (float)(DM - 1));
  const float inv  = 1.0f / (stdv + 1e-6f);
  _Float16* o = out + (size_t)row * DM;
  for (int i = lane; i < DM; i += 32)
    o[i] = (_Float16)(alpha[i] * (xr[i] - mean) * inv + beta[i]);
}

// ---------------- generic WMMA GEMM: C[M,N] = A[M,K] @ W[N,K]^T + bias -------
// The block's 16 A-rows are shared by all 4 waves: stage the 1KB A k-tile into
// LDS once per block with async-to-LDS DMA, double buffered against the WMMAs.
// mode 0: store f16        mode 1: relu -> f16
// mode 2: +resid -> f32    mode 3: f16, per-head transposed store [B,H,64,S]
__global__ __launch_bounds__(128) void wgemm_kernel(
    const _Float16* __restrict__ A, const _Float16* __restrict__ W,
    const float* __restrict__ bias, float* __restrict__ outf,
    _Float16* __restrict__ outh, const float* __restrict__ resid,
    int M, int N, int K, int mode) {
  __shared__ __align__(16) _Float16 atile[2][16 * 32];  // 2 x 1KB double buffer

  const int tid  = threadIdx.x;
  const int lane = tid & 31;
  const int wave = tid >> 5;
  const int hf   = lane >> 4;
  const int l16  = lane & 15;
  const int m0 = blockIdx.x * 16;
  const int n0 = blockIdx.y * 256 + wave * 64;

  v8f acc[4] = {};
  const _Float16* A0    = A + (size_t)m0 * K;           // block's 16 rows
  const _Float16* brow0 = W + (size_t)(n0 + l16) * K;

  // lanes 0..63 (waves 0,1) each DMA one 16B chunk: row = t>>1, half-row = t&1
  const int srow = tid >> 1, sseg = tid & 1;

  // prologue: stage k=0 into buffer 0
  if (tid < 64)
    async_cp_b128(A0 + (size_t)srow * K + sseg * 16,
                  &atile[0][srow * 32 + sseg * 16]);

  for (int k = 0; k < K; k += 32) {
    const int buf = (k >> 5) & 1;
    __syncthreads();  // prior reads of buf^1 done before DMA overwrites it
    if (k + 32 < K) {
      if (tid < 64)
        async_cp_b128(A0 + (size_t)srow * K + (k + 32) + sseg * 16,
                      &atile[buf ^ 1][srow * 32 + sseg * 16]);
      asm volatile("s_wait_asynccnt 1" ::: "memory");  // oldest (buf) landed
    } else {
      asm volatile("s_wait_asynccnt 0" ::: "memory");
    }
    __syncthreads();  // publish buf to all waves

    const v16h a = frag_ld_a(&atile[buf][l16 * 32], hf);
    __builtin_prefetch(brow0 + k + 128, 0, 1);  // global_prefetch_b8
#pragma unroll
    for (int t = 0; t < 4; ++t) {
      const v16h bf = frag_ld_b(brow0 + (size_t)t * 16 * K + k, hf);
      acc[t] = WMMA_F16(a, bf, acc[t]);
    }
  }

#pragma unroll
  for (int t = 0; t < 4; ++t) {
    const int col = n0 + t * 16 + l16;
    const float bv = bias[col];
#pragma unroll
    for (int r = 0; r < 8; ++r) {
      const int row = m0 + r + 8 * hf;   // C layout: vgpr r -> row r + 8*half
      float v = acc[t][r] + bv;
      if (mode == 1) v = v > 0.f ? v : 0.f;
      if (mode == 2) {
        outf[(size_t)row * N + col] = v + resid[(size_t)row * N + col];
      } else if (mode == 3) {
        const int b = row >> 10, s = row & (SSEQ - 1);
        const int h = col >> 6,  d = col & (DKH - 1);
        outh[(((size_t)b * NH + h) * DKH + d) * SSEQ + s] = (_Float16)v;
      } else {
        outh[(size_t)row * N + col] = (_Float16)v;
      }
    }
  }
}

// ---------------- fused flash attention: one wave = 16 queries of one (b,h) --
__global__ __launch_bounds__(128) void attn_kernel(
    const _Float16* __restrict__ Q,   // [B*S, 768]
    const _Float16* __restrict__ Km,  // [B*S, 768]
    const _Float16* __restrict__ Vt,  // [B,H,64,S]
    const int* __restrict__ mask,     // [B, S]
    _Float16* __restrict__ ctx) {     // [B*S, 768]
  __shared__ __align__(16) _Float16 plds[4][16 * 32];

  const int lane = threadIdx.x & 31;
  const int wave = threadIdx.x >> 5;
  const int hf   = lane >> 4;
  const int l16  = lane & 15;

  const int task = blockIdx.x * 4 + wave;  // B*H*(S/16) = 6144 tasks
  const int qt = task & 63;                // query tile
  const int bh = task >> 6;                // 0..95
  const int b = bh / NH, h = bh % NH;

  const size_t rowQ = ((size_t)b * SSEQ + qt * 16 + l16) * DM + h * DKH;
  const v16h aq0 = frag_ld_a(Q + rowQ, hf);       // d 0..31
  const v16h aq1 = frag_ld_a(Q + rowQ + 32, hf);  // d 32..63

  v8f o[4] = {};
  float mrow[8], lrow[8];
#pragma unroll
  for (int r = 0; r < 8; ++r) { mrow[r] = -1e30f; lrow[r] = 0.f; }

  const _Float16* Kbase = Km + (size_t)b * SSEQ * DM + h * DKH;
  const _Float16* Vbase = Vt + (size_t)bh * DKH * SSEQ;
  const int* mk = mask + b * SSEQ;
  _Float16* tile = plds[wave];

  for (int j = 0; j < SSEQ; j += 32) {
    // scores: two 16-key C tiles, K-dim = 64 -> 2 WMMAs each
    v8f s01[2];
#pragma unroll
    for (int c = 0; c < 2; ++c) {
      const int key = j + c * 16 + l16;
      const _Float16* krow = Kbase + (size_t)key * DM;
      v8f sv = {};
      sv = WMMA_F16(aq0, frag_ld_b(krow, hf), sv);
      sv = WMMA_F16(aq1, frag_ld_b(krow + 32, hf), sv);
      const bool keep = (mk[key] != 0);
#pragma unroll
      for (int r = 0; r < 8; ++r) sv[r] = keep ? sv[r] * 0.125f : 1e-9f;
      s01[c] = sv;
    }
    // online softmax per row (rows live in per-VGPR slots; reduce over 16-lane half)
#pragma unroll
    for (int r = 0; r < 8; ++r) {
      float cm = fmaxf(s01[0][r], s01[1][r]);
#pragma unroll
      for (int off = 8; off; off >>= 1) cm = fmaxf(cm, __shfl_xor(cm, off, 32));
      const float mnew = fmaxf(mrow[r], cm);
      const float fac  = __expf(mrow[r] - mnew);
      mrow[r] = mnew;
      const float p0 = __expf(s01[0][r] - mnew);
      const float p1 = __expf(s01[1][r] - mnew);
      s01[0][r] = p0; s01[1][r] = p1;
      float ps = p0 + p1;
#pragma unroll
      for (int off = 8; off; off >>= 1) ps += __shfl_xor(ps, off, 32);
      lrow[r] = lrow[r] * fac + ps;
#pragma unroll
      for (int t = 0; t < 4; ++t) o[t][r] *= fac;
    }
    // C-layout -> A-layout via 1KB LDS bounce (same-wave, DS in-order + wait)
#pragma unroll
    for (int r = 0; r < 8; ++r) {
      const int rr = r + 8 * hf;
      tile[rr * 32 + l16]      = (_Float16)s01[0][r];
      tile[rr * 32 + 16 + l16] = (_Float16)s01[1][r];
    }
    asm volatile("s_wait_dscnt 0" ::: "memory");
    const v16h pa = frag_ld_a(tile + l16 * 32, hf);
    // ctx += P(16x32) @ V(32x64)
#pragma unroll
    for (int t = 0; t < 4; ++t) {
      const _Float16* vr = Vbase + (size_t)(t * 16 + l16) * SSEQ + j;
      o[t] = WMMA_F16(pa, frag_ld_b(vr, hf), o[t]);
    }
  }

#pragma unroll
  for (int t = 0; t < 4; ++t) {
#pragma unroll
    for (int r = 0; r < 8; ++r) {
      const int srow = qt * 16 + r + 8 * hf;
      const float v = o[t][r] / lrow[r];
      ctx[((size_t)b * SSEQ + srow) * DM + h * DKH + t * 16 + l16] = (_Float16)v;
    }
  }
}

extern "C" void kernel_launch(void* const* d_in, const int* in_sizes, int n_in,
                              void* d_out, int out_size, void* d_ws, size_t ws_size,
                              hipStream_t stream) {
  (void)in_sizes; (void)n_in; (void)out_size; (void)ws_size;
  const float* x     = (const float*)d_in[0];
  const int*   mask  = (const int*)d_in[1];
  const float* wq    = (const float*)d_in[2];
  const float* bq    = (const float*)d_in[3];
  const float* wk    = (const float*)d_in[4];
  const float* bk    = (const float*)d_in[5];
  const float* wv    = (const float*)d_in[6];
  const float* bv    = (const float*)d_in[7];
  const float* wo    = (const float*)d_in[8];
  const float* bo    = (const float*)d_in[9];
  const float* ln1a  = (const float*)d_in[10];
  const float* ln1b  = (const float*)d_in[11];
  const float* ln2a  = (const float*)d_in[12];
  const float* ln2b  = (const float*)d_in[13];
  const float* w1    = (const float*)d_in[14];
  const float* b1    = (const float*)d_in[15];
  const float* w2    = (const float*)d_in[16];
  const float* b2    = (const float*)d_in[17];
  float* out = (float*)d_out;

  char* ws = (char*)d_ws;
  size_t off = 0;
  auto alloc = [&](size_t bytes) -> void* {
    void* p = ws + off;
    off += (bytes + 255) & ~(size_t)255;
    return p;
  };
  _Float16* wq16  = (_Float16*)alloc((size_t)DM * DM * 2);
  _Float16* wk16  = (_Float16*)alloc((size_t)DM * DM * 2);
  _Float16* wv16  = (_Float16*)alloc((size_t)DM * DM * 2);
  _Float16* wo16  = (_Float16*)alloc((size_t)DM * DM * 2);
  _Float16* w116  = (_Float16*)alloc((size_t)DFF * DM * 2);
  _Float16* w216  = (_Float16*)alloc((size_t)DM * DFF * 2);
  _Float16* xn1   = (_Float16*)alloc((size_t)MROWS * DM * 2);
  _Float16* q16   = (_Float16*)alloc((size_t)MROWS * DM * 2);
  _Float16* k16   = (_Float16*)alloc((size_t)MROWS * DM * 2);
  _Float16* vt16  = (_Float16*)alloc((size_t)MROWS * DM * 2);
  _Float16* ctx16 = (_Float16*)alloc((size_t)MROWS * DM * 2);
  float*    res1  = (float*)   alloc((size_t)MROWS * DM * 4);
  _Float16* xn2   = (_Float16*)alloc((size_t)MROWS * DM * 2);
  _Float16* h16   = (_Float16*)alloc((size_t)MROWS * DFF * 2);

  auto cvt = [&](const float* src, _Float16* dst, int n) {
    cvt_f32_f16_kernel<<<(n + 255) / 256, 256, 0, stream>>>(src, dst, n);
  };
  cvt(wq, wq16, DM * DM);
  cvt(wk, wk16, DM * DM);
  cvt(wv, wv16, DM * DM);
  cvt(wo, wo16, DM * DM);
  cvt(w1, w116, DFF * DM);
  cvt(w2, w216, DM * DFF);

  // LN1
  ln_kernel<<<MROWS / 8, 256, 0, stream>>>(x, ln1a, ln1b, xn1, MROWS);

  const dim3 blk(128);
  // Q, K projections (f16 out), V projection (per-head transposed f16 out)
  wgemm_kernel<<<dim3(MROWS / 16, DM / 256), blk, 0, stream>>>(
      xn1, wq16, bq, nullptr, q16, nullptr, MROWS, DM, DM, 0);
  wgemm_kernel<<<dim3(MROWS / 16, DM / 256), blk, 0, stream>>>(
      xn1, wk16, bk, nullptr, k16, nullptr, MROWS, DM, DM, 0);
  wgemm_kernel<<<dim3(MROWS / 16, DM / 256), blk, 0, stream>>>(
      xn1, wv16, bv, nullptr, vt16, nullptr, MROWS, DM, DM, 3);

  // fused attention: B*H*(S/16) = 6144 wave-tasks, 4 waves/block
  attn_kernel<<<dim3(6144 / 4), blk, 0, stream>>>(q16, k16, vt16, mask, ctx16);

  // output projection + residual -> f32
  wgemm_kernel<<<dim3(MROWS / 16, DM / 256), blk, 0, stream>>>(
      ctx16, wo16, bo, res1, nullptr, x, MROWS, DM, DM, 2);

  // LN2
  ln_kernel<<<MROWS / 8, 256, 0, stream>>>(res1, ln2a, ln2b, xn2, MROWS);

  // FFN1 (ReLU) and FFN2 (+residual -> d_out)
  wgemm_kernel<<<dim3(MROWS / 16, DFF / 256), blk, 0, stream>>>(
      xn2, w116, b1, nullptr, h16, nullptr, MROWS, DFF, DM, 1);
  wgemm_kernel<<<dim3(MROWS / 16, DM / 256), blk, 0, stream>>>(
      h16, w216, b2, out, nullptr, res1, MROWS, DM, DFF, 2);
}
